// Retention_86328842650435
// MI455X (gfx1250) — matrix operand order repaired and checked by reading, not compiled, not run on hardware
//
#include <hip/hip_runtime.h>
#include <hip/hip_bf16.h>
#include <math.h>

// Problem constants (B=1)
constexpr int kT  = 2048;
constexpr int kE  = 1024;
constexpr int kH  = 16;
constexpr int kHD = 64;

typedef __attribute__((ext_vector_type(16))) _Float16 v16h;
typedef __attribute__((ext_vector_type(8)))  _Float16 v8h;
typedef __attribute__((ext_vector_type(8)))  float    v8f;

union AFrag { v16h v; v8h h[2]; };

// ---------------------------------------------------------------------------
// f32 -> f16 conversion
// ---------------------------------------------------------------------------
__global__ void cvt_f32_f16(const float* __restrict__ src,
                            _Float16* __restrict__ dst, int n) {
  int i = blockIdx.x * blockDim.x + threadIdx.x;
  if (i < n) dst[i] = (_Float16)src[i];
}

// ---------------------------------------------------------------------------
// WMMA GEMM: C[M][N] = A[M][K] @ W[N][K]^T + bias[N]   (f16 in, f32 out)
// Block = 128 threads (4 waves). Block tile 128x64; each wave does 32x64
// (two A fragments share every B fragment -> 2 wmma per B load, double the
// arithmetic intensity of a 16-row tile). Fragments loaded straight from
// global in a contiguous-k layout (same k order for A and B):
//   A (16x32 f16): lane l -> row (l&15); 16 contiguous halves at K = 16*(l>>4)
//   B (32x16 f16): lane l -> col (l&15); 16 contiguous halves at K = 16*(l>>4)
// Both are contiguous because the contraction dim is innermost for x and W.
// ---------------------------------------------------------------------------
__global__ __launch_bounds__(128) void gemm_f16_wt(
    const _Float16* __restrict__ A, const _Float16* __restrict__ W,
    const float* __restrict__ bias, float* __restrict__ C,
    int M, int N, int K) {
  const int wave = threadIdx.x >> 5;
  const int lane = threadIdx.x & 31;
  const int ln = lane & 15;
  const int hl = lane >> 4;
  const int rowBase = blockIdx.x * 128 + wave * 32;
  const int colBase = blockIdx.y * 64;

  const _Float16* __restrict__ arow0 = A + (size_t)(rowBase + ln) * K;
  const _Float16* __restrict__ arow1 = arow0 + (size_t)16 * K;
  const _Float16* __restrict__ wrow  = W + (size_t)(colBase + ln) * K;

  v8f acc[8] = {};  // [0..3] rows 0..15, [4..7] rows 16..31

  for (int kk = 0; kk < K; kk += 32) {
    AFrag a0, a1;
    a0.h[0] = *(const v8h*)(arow0 + kk + 16 * hl);
    a0.h[1] = *(const v8h*)(arow0 + kk + 16 * hl + 8);
    a1.h[0] = *(const v8h*)(arow1 + kk + 16 * hl);
    a1.h[1] = *(const v8h*)(arow1 + kk + 16 * hl + 8);
    __builtin_prefetch(arow0 + kk + 512, 0, 3);
    __builtin_prefetch(arow1 + kk + 512, 0, 3);
    __builtin_prefetch(wrow + kk + 512, 0, 3);
#pragma unroll
    for (int nt = 0; nt < 4; ++nt) {
      v16h b = *(const v16h*)(wrow + (size_t)nt * 16 * K + kk + 16 * hl);
      acc[nt] = __builtin_amdgcn_wmma_f32_16x16x32_f16(
          false, a0.v, false, b, (short)0, acc[nt], false, false);
      acc[4 + nt] = __builtin_amdgcn_wmma_f32_16x16x32_f16(
          false, a1.v, false, b, (short)0, acc[4 + nt], false, false);
    }
  }

#pragma unroll
  for (int rt = 0; rt < 2; ++rt) {
#pragma unroll
    for (int nt = 0; nt < 4; ++nt) {
#pragma unroll
      for (int r = 0; r < 8; ++r) {
        int row = rowBase + rt * 16 + r + 8 * hl;
        int col = colBase + nt * 16 + ln;
        C[(size_t)row * N + col] = acc[rt * 4 + nt][r] + bias[col];
      }
    }
  }
}

// ---------------------------------------------------------------------------
// RoPE + pack: Qraw/Kraw/Vraw [T][E] f32  ->  Qh,Kh [H][T][HD] f16 (Q scaled
// by 1/sqrt(HD)), Vt [H][HD][T] f16 (transposed so V B-fragments are
// contiguous in time).
// ---------------------------------------------------------------------------
__global__ void rope_pack(const float* __restrict__ Qr,
                          const float* __restrict__ Kr,
                          const float* __restrict__ Vr,
                          _Float16* __restrict__ Qh,
                          _Float16* __restrict__ Kh,
                          _Float16* __restrict__ Vt) {
  int idx = blockIdx.x * blockDim.x + threadIdx.x;  // T*H*(HD/2) threads
  int i = idx & 31;          // pair index within head dim
  int h = (idx >> 5) & 15;
  int t = idx >> 9;
  int d0 = 2 * i, d1 = d0 + 1;
  int j0 = d0 & 31, j1 = d1 & 31;  // emb = concat(freqs, freqs)

  const float kLn1e4o32 = 0.28782313662425572f;  // ln(10000)/32
  float f0 = __expf(-(float)j0 * kLn1e4o32);
  float f1 = __expf(-(float)j1 * kLn1e4o32);
  float a0 = (float)t * f0, a1 = (float)t * f1;
  float c0 = __cosf(a0), s0 = __sinf(a0);
  float c1 = __cosf(a1), s1 = __sinf(a1);

  size_t src = (size_t)t * kE + h * kHD;
  float q0 = Qr[src + d0], q1 = Qr[src + d1];
  float k0 = Kr[src + d0], k1 = Kr[src + d1];
  // out[2i] = x[2i]*cos[2i] - x[2i+1]*sin[2i]; out[2i+1] = x[2i+1]*cos + x[2i]*sin
  float qo0 = q0 * c0 - q1 * s0;
  float qo1 = q1 * c1 + q0 * s1;
  float ko0 = k0 * c0 - k1 * s0;
  float ko1 = k1 * c1 + k0 * s1;

  size_t dst = ((size_t)h * kT + t) * kHD;
  Qh[dst + d0] = (_Float16)(qo0 * 0.125f);  // fold 1/sqrt(64)
  Qh[dst + d1] = (_Float16)(qo1 * 0.125f);
  Kh[dst + d0] = (_Float16)ko0;
  Kh[dst + d1] = (_Float16)ko1;

  Vt[((size_t)h * kHD + d0) * kT + t] = (_Float16)Vr[src + d0];
  Vt[((size_t)h * kHD + d1) * kT + t] = (_Float16)Vr[src + d1];
}

// ---------------------------------------------------------------------------
// Column mean of x over T
// ---------------------------------------------------------------------------
__global__ void colmean_kernel(const float* __restrict__ x,
                               float* __restrict__ xmean) {
  int col = blockIdx.x * blockDim.x + threadIdx.x;
  if (col >= kE) return;
  float s = 0.0f;
  for (int t = 0; t < kT; ++t) s += x[(size_t)t * kE + col];
  xmean[col] = s * (1.0f / (float)kT);
}

// ---------------------------------------------------------------------------
// log_decay[h] = log(1 - 2^(-5 - 8*sigmoid(xmean . wd[h] + bd[h])))
// ---------------------------------------------------------------------------
__global__ void decay_kernel(const float* __restrict__ xmean,
                             const float* __restrict__ wd,
                             const float* __restrict__ bd,
                             float* __restrict__ ld) {
  __shared__ float red[256];
  int h = blockIdx.x;
  float p = 0.0f;
  for (int e = threadIdx.x; e < kE; e += 256)
    p += xmean[e] * wd[(size_t)h * kE + e];
  red[threadIdx.x] = p;
  __syncthreads();
  for (int st = 128; st > 0; st >>= 1) {
    if (threadIdx.x < st) red[threadIdx.x] += red[threadIdx.x + st];
    __syncthreads();
  }
  if (threadIdx.x == 0) {
    float pd = 1.0f / (1.0f + __expf(-(red[0] + bd[h])));
    ld[h] = __logf(1.0f - exp2f(-5.0f - 8.0f * pd));
  }
}

// ---------------------------------------------------------------------------
// Flash-style retention attention. Grid (H, T/64), block = 128 (4 waves);
// each wave owns one 16-row q tile and streams k in steps of 32.
//   S  = Q K^T  : 2 chained WMMAs per 16x16 tile (head dim K=64)
//   bias        : max(k-q,0) * log_decay[h]  (scale folded into Q)
//   softmax     : online, half-wave (__shfl_xor width 16) row reductions
//   O += P V    : P staged via LDS (C-layout -> A-layout), 4 WMMAs vs Vt
// Writes O [T][E] f32.
// ---------------------------------------------------------------------------
__global__ __launch_bounds__(128) void attn_kernel(
    const _Float16* __restrict__ Qh, const _Float16* __restrict__ Kh,
    const _Float16* __restrict__ Vt, const float* __restrict__ logdecay,
    float* __restrict__ O) {
  __shared__ _Float16 pbuf[4][16 * 32];

  const int h    = blockIdx.x;
  const int wave = threadIdx.x >> 5;
  const int lane = threadIdx.x & 31;
  const int ln = lane & 15;
  const int hl = lane >> 4;
  const int qbase = (blockIdx.y * 4 + wave) * 16;
  const float ld = logdecay[h];

  // Q A-fragments for the whole tile (head dim split 0..31 / 32..63)
  const _Float16* __restrict__ qrow =
      Qh + ((size_t)h * kT + qbase + ln) * kHD;
  AFrag aq0, aq1;
  aq0.h[0] = *(const v8h*)(qrow + 16 * hl);
  aq0.h[1] = *(const v8h*)(qrow + 16 * hl + 8);
  aq1.h[0] = *(const v8h*)(qrow + 32 + 16 * hl);
  aq1.h[1] = *(const v8h*)(qrow + 40 + 16 * hl);

  v8f oacc[4] = {};
  float m[8], s[8];
#pragma unroll
  for (int r = 0; r < 8; ++r) { m[r] = -1e30f; s[r] = 0.0f; }

  _Float16* myp = pbuf[wave];

  for (int kt = 0; kt < kT; kt += 32) {
    // --- S = Q K^T for 32 k columns (two 16x16 tiles) ---
    const _Float16* __restrict__ krow0 =
        Kh + ((size_t)h * kT + kt + ln) * kHD;
    const _Float16* __restrict__ krow1 = krow0 + 16 * kHD;
    v16h bk0lo = *(const v16h*)(krow0 + 16 * hl);
    v16h bk0hi = *(const v16h*)(krow0 + 32 + 16 * hl);
    v16h bk1lo = *(const v16h*)(krow1 + 16 * hl);
    v16h bk1hi = *(const v16h*)(krow1 + 32 + 16 * hl);

    v8f s0 = {}, s1 = {};
    s0 = __builtin_amdgcn_wmma_f32_16x16x32_f16(false, aq0.v, false, bk0lo,
                                                (short)0, s0, false, false);
    s0 = __builtin_amdgcn_wmma_f32_16x16x32_f16(false, aq1.v, false, bk0hi,
                                                (short)0, s0, false, false);
    s1 = __builtin_amdgcn_wmma_f32_16x16x32_f16(false, aq0.v, false, bk1lo,
                                                (short)0, s1, false, false);
    s1 = __builtin_amdgcn_wmma_f32_16x16x32_f16(false, aq1.v, false, bk1hi,
                                                (short)0, s1, false, false);

    // --- decay bias + online softmax (rows live across one 16-lane half) ---
    float p0[8], p1[8];
#pragma unroll
    for (int r = 0; r < 8; ++r) {
      int q  = qbase + r + 8 * hl;
      int k0 = kt + ln;
      int k1 = k0 + 16;
      p0[r] = s0[r] + ((k0 > q) ? (float)(k0 - q) * ld : 0.0f);
      p1[r] = s1[r] + ((k1 > q) ? (float)(k1 - q) * ld : 0.0f);
    }
#pragma unroll
    for (int r = 0; r < 8; ++r) {
      float mx = fmaxf(p0[r], p1[r]);
      mx = fmaxf(mx, __shfl_xor(mx, 1, 16));
      mx = fmaxf(mx, __shfl_xor(mx, 2, 16));
      mx = fmaxf(mx, __shfl_xor(mx, 4, 16));
      mx = fmaxf(mx, __shfl_xor(mx, 8, 16));
      float nm    = fmaxf(m[r], mx);
      float alpha = __expf(m[r] - nm);
      m[r] = nm;
      p0[r] = __expf(p0[r] - nm);
      p1[r] = __expf(p1[r] - nm);
      float rs = p0[r] + p1[r];
      rs += __shfl_xor(rs, 1, 16);
      rs += __shfl_xor(rs, 2, 16);
      rs += __shfl_xor(rs, 4, 16);
      rs += __shfl_xor(rs, 8, 16);
      s[r] = s[r] * alpha + rs;
#pragma unroll
      for (int nt = 0; nt < 4; ++nt) oacc[nt][r] *= alpha;
    }

    // --- stage P (C-layout) into LDS, reload as A-fragment ---
#pragma unroll
    for (int r = 0; r < 8; ++r) {
      int row = r + 8 * hl;
      myp[row * 32 + ln]      = (_Float16)p0[r];
      myp[row * 32 + 16 + ln] = (_Float16)p1[r];
    }
    __syncthreads();
    AFrag pf;
    pf.h[0] = *(const v8h*)(myp + ln * 32 + 16 * hl);
    pf.h[1] = *(const v8h*)(myp + ln * 32 + 16 * hl + 8);

    // --- O += P V : V B-fragments contiguous in time thanks to Vt layout ---
#pragma unroll
    for (int nt = 0; nt < 4; ++nt) {
      const _Float16* __restrict__ vcol =
          Vt + ((size_t)h * kHD + nt * 16 + ln) * kT + kt + 16 * hl;
      v16h bv = *(const v16h*)vcol;
      oacc[nt] = __builtin_amdgcn_wmma_f32_16x16x32_f16(
          false, pf.v, false, bv, (short)0, oacc[nt], false, false);
    }
    __syncthreads();
  }

  // --- normalize and store O as [T][E] f32 ---
#pragma unroll
  for (int r = 0; r < 8; ++r) {
    float inv = 1.0f / s[r];
    int q = qbase + r + 8 * hl;
#pragma unroll
    for (int nt = 0; nt < 4; ++nt) {
      int d = nt * 16 + ln;
      O[(size_t)q * kE + h * kHD + d] = oacc[nt][r] * inv;
    }
  }
}

// ---------------------------------------------------------------------------
// Per-head group-norm stats over (T, HD)
// ---------------------------------------------------------------------------
__global__ void gnstats_kernel(const float* __restrict__ O,
                               float* __restrict__ mu,
                               float* __restrict__ var) {
  __shared__ float rs[256], rq[256];
  int h = blockIdx.x;
  float s = 0.0f, q = 0.0f;
  for (int i = threadIdx.x; i < kT * kHD; i += 256) {
    int t = i >> 6, d = i & 63;
    float v = O[(size_t)t * kE + h * kHD + d];
    s += v;
    q += v * v;
  }
  rs[threadIdx.x] = s;
  rq[threadIdx.x] = q;
  __syncthreads();
  for (int st = 128; st > 0; st >>= 1) {
    if (threadIdx.x < st) {
      rs[threadIdx.x] += rs[threadIdx.x + st];
      rq[threadIdx.x] += rq[threadIdx.x + st];
    }
    __syncthreads();
  }
  if (threadIdx.x == 0) {
    const float n = (float)(kT * kHD);
    float mn = rs[0] / n;
    mu[h]  = mn;
    var[h] = rq[0] / n - mn * mn;
  }
}

// ---------------------------------------------------------------------------
// A16 = f16( silu(G) * ((O - mu)/sqrt(var+eps)*gamma + beta) )
// ---------------------------------------------------------------------------
__global__ void gate_mix_kernel(const float* __restrict__ O,
                                const float* __restrict__ G,
                                const float* __restrict__ mu,
                                const float* __restrict__ var,
                                const float* __restrict__ gamma,
                                const float* __restrict__ beta,
                                _Float16* __restrict__ A16) {
  int idx = blockIdx.x * blockDim.x + threadIdx.x;  // T*E threads
  int e = idx & (kE - 1);
  int h = e >> 6;
  float on = (O[idx] - mu[h]) * rsqrtf(var[h] + 1e-5f) * gamma[e] + beta[e];
  float g = G[idx];
  float gate = g / (1.0f + __expf(-g));  // silu
  A16[idx] = (_Float16)(gate * on);
}

// ---------------------------------------------------------------------------
// Host launcher
// ---------------------------------------------------------------------------
extern "C" void kernel_launch(void* const* d_in, const int* in_sizes, int n_in,
                              void* d_out, int out_size, void* d_ws,
                              size_t ws_size, hipStream_t stream) {
  (void)in_sizes; (void)n_in; (void)out_size; (void)ws_size;

  const float* x     = (const float*)d_in[0];
  const float* wq    = (const float*)d_in[1];
  const float* bq    = (const float*)d_in[2];
  const float* wk    = (const float*)d_in[3];
  const float* bk    = (const float*)d_in[4];
  const float* wv    = (const float*)d_in[5];
  const float* bv    = (const float*)d_in[6];
  const float* wg    = (const float*)d_in[7];
  const float* bg    = (const float*)d_in[8];
  const float* wo    = (const float*)d_in[9];
  const float* bo    = (const float*)d_in[10];
  const float* wd    = (const float*)d_in[11];
  const float* bd    = (const float*)d_in[12];
  const float* gamma = (const float*)d_in[13];
  const float* beta  = (const float*)d_in[14];
  float* out = (float*)d_out;

  char* ws = (char*)d_ws;
  size_t off = 0;
  auto alloc = [&](size_t bytes) {
    off = (off + 255) & ~(size_t)255;
    size_t o = off;
    off += bytes;
    return o;
  };

  constexpr size_t nTE = (size_t)kT * kE;  // 2,097,152
  constexpr size_t nEE = (size_t)kE * kE;  // 1,048,576

  const size_t o_x16  = alloc(nTE * 2);
  const size_t o_wq16 = alloc(nEE * 2);
  const size_t o_wk16 = alloc(nEE * 2);
  const size_t o_wv16 = alloc(nEE * 2);
  const size_t o_wg16 = alloc(nEE * 2);
  const size_t o_wo16 = alloc(nEE * 2);
  const size_t o_Qraw = alloc(nTE * 4);
  const size_t o_Kraw = alloc(nTE * 4);
  const size_t o_Vraw = alloc(nTE * 4);
  const size_t o_Graw = alloc(nTE * 4);
  const size_t o_Qh   = alloc(nTE * 2);
  const size_t o_Kh   = alloc(nTE * 2);
  const size_t o_Vt   = alloc(nTE * 2);
  const size_t o_xm   = alloc(kE * 4);
  const size_t o_ld   = alloc(kH * 4);
  const size_t o_mu   = alloc(kH * 4);
  const size_t o_var  = alloc(kH * 4);
  // Aliases over buffers that are dead by the time these are written:
  const size_t o_O    = o_Qraw;  // O written after rope_pack consumed Qraw
  const size_t o_A16  = o_Kraw;  // A16 written after rope_pack consumed Kraw

  _Float16* x16  = (_Float16*)(ws + o_x16);
  _Float16* wq16 = (_Float16*)(ws + o_wq16);
  _Float16* wk16 = (_Float16*)(ws + o_wk16);
  _Float16* wv16 = (_Float16*)(ws + o_wv16);
  _Float16* wg16 = (_Float16*)(ws + o_wg16);
  _Float16* wo16 = (_Float16*)(ws + o_wo16);
  float*    Qraw = (float*)(ws + o_Qraw);
  float*    Kraw = (float*)(ws + o_Kraw);
  float*    Vraw = (float*)(ws + o_Vraw);
  float*    Graw = (float*)(ws + o_Graw);
  _Float16* Qh   = (_Float16*)(ws + o_Qh);
  _Float16* Kh   = (_Float16*)(ws + o_Kh);
  _Float16* Vt   = (_Float16*)(ws + o_Vt);
  float*    xm   = (float*)(ws + o_xm);
  float*    ldv  = (float*)(ws + o_ld);
  float*    mu   = (float*)(ws + o_mu);
  float*    var  = (float*)(ws + o_var);
  float*    O    = (float*)(ws + o_O);
  _Float16* A16  = (_Float16*)(ws + o_A16);

  // 1) f16 conversions
  cvt_f32_f16<<<(int)((nTE + 255) / 256), 256, 0, stream>>>(x, x16, (int)nTE);
  cvt_f32_f16<<<(int)((nEE + 255) / 256), 256, 0, stream>>>(wq, wq16, (int)nEE);
  cvt_f32_f16<<<(int)((nEE + 255) / 256), 256, 0, stream>>>(wk, wk16, (int)nEE);
  cvt_f32_f16<<<(int)((nEE + 255) / 256), 256, 0, stream>>>(wv, wv16, (int)nEE);
  cvt_f32_f16<<<(int)((nEE + 255) / 256), 256, 0, stream>>>(wg, wg16, (int)nEE);
  cvt_f32_f16<<<(int)((nEE + 255) / 256), 256, 0, stream>>>(wo, wo16, (int)nEE);

  // 2) Q/K/V/G projections via WMMA (block tile 128x64, wave tile 32x64)
  dim3 ggrid(kT / 128, kE / 64);
  gemm_f16_wt<<<ggrid, 128, 0, stream>>>(x16, wq16, bq, Qraw, kT, kE, kE);
  gemm_f16_wt<<<ggrid, 128, 0, stream>>>(x16, wk16, bk, Kraw, kT, kE, kE);
  gemm_f16_wt<<<ggrid, 128, 0, stream>>>(x16, wv16, bv, Vraw, kT, kE, kE);
  gemm_f16_wt<<<ggrid, 128, 0, stream>>>(x16, wg16, bg, Graw, kT, kE, kE);

  // 3) decay gates
  colmean_kernel<<<kE / 256, 256, 0, stream>>>(x, xm);
  decay_kernel<<<kH, 256, 0, stream>>>(xm, wd, bd, ldv);

  // 4) RoPE + pack into WMMA-friendly layouts
  rope_pack<<<(kT * kH * (kHD / 2)) / 256, 256, 0, stream>>>(Qraw, Kraw, Vraw,
                                                             Qh, Kh, Vt);

  // 5) flash-style retention attention (WMMA)
  attn_kernel<<<dim3(kH, kT / 64), 128, 0, stream>>>(Qh, Kh, Vt, ldv, O);

  // 6) per-head group norm stats
  gnstats_kernel<<<kH, 256, 0, stream>>>(O, mu, var);

  // 7) norm + SiLU gate -> f16 A operand
  gate_mix_kernel<<<(int)(nTE / 256), 256, 0, stream>>>(O, Graw, mu, var,
                                                        gamma, beta, A16);

  // 8) output projection via WMMA
  gemm_f16_wt<<<ggrid, 128, 0, stream>>>(A16, wo16, bo, out, kT, kE, kE);
}
